// PECEngine_27625229648149
// MI455X (gfx1250) — matrix operand order repaired and checked by reading, not compiled, not run on hardware
//
#include <hip/hip_runtime.h>
#include <hip/hip_bf16.h>

typedef _Float16 v16h __attribute__((ext_vector_type(16)));
typedef _Float16 h8v  __attribute__((ext_vector_type(8)));
typedef float    v8f  __attribute__((ext_vector_type(8)));
typedef int      i32x4 __attribute__((vector_size(16)));

#if defined(__has_builtin)
#if __has_builtin(__builtin_amdgcn_global_load_async_to_lds_b128)
#define HAVE_ASYNC_COPY 1
#endif
#endif

// Problem constants
constexpr int Bb  = 4;
constexpr int Qn  = 2048;
constexpr int Dm  = 2048;
constexpr int Hh  = 16;
constexpr int KVh = 8;
constexpr int HD  = 128;
constexpr int Mm  = 64;
constexpr int QKVW = Hh*HD + 2*KVh*HD;  // 4096
constexpr int KTOT = Mm + Qn;           // 2112 (multiple of 32)
constexpr float QSCALE = 0.08838834764831845f; // 1/sqrt(128)

#define CAT16(a, b) __builtin_shufflevector(a, b, 0,1,2,3,4,5,6,7,8,9,10,11,12,13,14,15)

// ---------------------------------------------------------------------------
// elementwise f32 -> f16
__global__ void k_cvt_f16(const float* __restrict__ src, _Float16* __restrict__ dst, int n) {
    int i = blockIdx.x * blockDim.x + threadIdx.x;
    if (i < n) dst[i] = (_Float16)src[i];
}

// pack [Wq | Wk | Wv] (row = D dim) into one f16 weight matrix 2048 x 4096
__global__ void k_pack_wqkv(const float* __restrict__ Wq, const float* __restrict__ Wk,
                            const float* __restrict__ Wv, _Float16* __restrict__ dst) {
    int i = blockIdx.x * blockDim.x + threadIdx.x;   // over 2048*4096
    int d = i / QKVW;
    int j = i % QKVW;
    float v;
    if (j < Hh*HD)               v = Wq[d*(Hh*HD) + j];
    else if (j < Hh*HD + KVh*HD) v = Wk[d*(KVh*HD) + (j - Hh*HD)];
    else                         v = Wv[d*(KVh*HD) + (j - Hh*HD - KVh*HD)];
    dst[i] = (_Float16)v;
}

// ---------------------------------------------------------------------------
// Tiled f16 x f16 -> f32 GEMM, compile-time N/K (all addressing = shifts).
// C[M,N] = A[M,K] * B[K,N], row-major.  Block: 256 threads = 8 waves.
// Tile 128x128, K-stage 64 (two 32-deep WMMA steps, 16 WMMAs per barrier).
// Software-pipelined: next tile's 8 global h8v loads are issued before the
// current tile's WMMAs, then dumped to LDS after the trailing barrier.
template<int N, int K>
__global__ void __launch_bounds__(256)
k_gemm_f16(const _Float16* __restrict__ A, const _Float16* __restrict__ Bm,
           float* __restrict__ C) {
    constexpr int BM = 128, BN = 128, BK = 64;
    __shared__ _Float16 At[BM * BK];     // 16 KB
    __shared__ v16h    BtV[4 * BN];      // 16 KB  [kgroup (k>>4)][col] -> 16 halves

    const int tid  = threadIdx.x;
    const int wid  = tid >> 5;
    const int lane = tid & 31;
    const int ln   = lane & 15;
    const int hi   = lane >> 4;
    const int wrow = (wid >> 2) * 64;    // 0 or 64
    const int wcol = (wid & 3) * 32;     // 0,32,64,96
    const int m0 = blockIdx.y * BM;
    const int n0 = blockIdx.x * BN;

    // staging coordinates (4 h8v vectors each for A and B per thread)
    int ar[4], ac[4], br[4], bc[4];
    for (int i = 0; i < 4; ++i) {
        int v = tid + 256 * i;
        ar[i] = v >> 3;  ac[i] = (v & 7) * 8;    // A: 8 vectors per 64-wide row
        br[i] = v >> 4;  bc[i] = (v & 15) * 8;   // B: 16 vectors per 128-wide row
    }

    v8f acc[4][2];
    for (int mt = 0; mt < 4; ++mt)
        for (int nt = 0; nt < 2; ++nt)
            for (int j = 0; j < 8; ++j) acc[mt][nt][j] = 0.0f;

    h8v aReg[4], bReg[4];
    auto loadTile = [&](int k0) {
        for (int i = 0; i < 4; ++i)
            aReg[i] = *(const h8v*)&A[(size_t)(m0 + ar[i]) * K + k0 + ac[i]];
        for (int i = 0; i < 4; ++i)
            bReg[i] = *(const h8v*)&Bm[(size_t)(k0 + br[i]) * N + n0 + bc[i]];
    };
    auto storeTile = [&]() {
        for (int i = 0; i < 4; ++i)
            *(h8v*)&At[ar[i] * BK + ac[i]] = aReg[i];
        for (int i = 0; i < 4; ++i) {
            _Float16* bp = (_Float16*)&BtV[(br[i] >> 4) * BN];
            int kk = br[i] & 15;
            for (int u = 0; u < 8; ++u)
                bp[(bc[i] + u) * 16 + kk] = bReg[i][u];
        }
    };

    loadTile(0);
    storeTile();
    for (int k0 = 0; k0 < K; k0 += BK) {
        __syncthreads();                         // LDS tile ready
        const bool more = (k0 + BK) < K;
        if (more) loadTile(k0 + BK);             // in flight under the WMMAs

        for (int kk = 0; kk < BK; kk += 32) {
            v16h af[4], bf[2];
            for (int mt = 0; mt < 4; ++mt) {
                const _Float16* ap = &At[(wrow + mt * 16 + ln) * BK + kk + 16 * hi];
                af[mt] = CAT16(*(const h8v*)ap, *(const h8v*)(ap + 8));
            }
            for (int nt = 0; nt < 2; ++nt)
                bf[nt] = BtV[((kk >> 4) + hi) * BN + wcol + nt * 16 + ln];
            for (int mt = 0; mt < 4; ++mt)
                for (int nt = 0; nt < 2; ++nt)
                    acc[mt][nt] = __builtin_amdgcn_wmma_f32_16x16x32_f16(
                        false, af[mt], false, bf[nt], (short)0, acc[mt][nt], false, false);
        }
        __syncthreads();                         // done reading LDS
        if (more) storeTile();                   // dump next tile
    }

    for (int mt = 0; mt < 4; ++mt)
        for (int nt = 0; nt < 2; ++nt)
            for (int j = 0; j < 8; ++j) {
                int row = m0 + wrow + mt * 16 + j + 8 * hi;   // C/D layout
                int col = n0 + wcol + nt * 16 + ln;
                C[(size_t)row * N + col] = acc[mt][nt][j];
            }
}

// ---------------------------------------------------------------------------
// RMSNorm + RoPE for Q and K heads.  grid.x = B*Q, grid.y = H + KV, block=128.
__global__ void __launch_bounds__(128)
k_norm_rope(const float* __restrict__ qkv, const float* __restrict__ cosb,
            const float* __restrict__ sinb, const float* __restrict__ qw,
            const float* __restrict__ kw, _Float16* __restrict__ qf,
            _Float16* __restrict__ kf) {
    const int row  = blockIdx.x;        // b*Q + qpos
    const int b    = row / Qn;
    const int qpos = row % Qn;
    const int hidx = blockIdx.y;
    const int t    = threadIdx.x;
    const bool isq = hidx < Hh;

    const float* base = isq ? (qkv + (size_t)row * QKVW + hidx * HD)
                            : (qkv + (size_t)row * QKVW + Hh * HD + (hidx - Hh) * HD);
    const float* w = isq ? qw : kw;

    float x = base[t];
    __shared__ float red[128];
    red[t] = x * x;
    __syncthreads();
    for (int s = 64; s > 0; s >>= 1) {
        if (t < s) red[t] += red[t + s];
        __syncthreads();
    }
    float inv = rsqrtf(red[0] * (1.0f / HD) + 1e-6f);

    int p = (t < 64) ? t + 64 : t - 64;
    float xp  = base[p];
    float xn  = x  * inv * w[t];
    float xpn = xp * inv * w[p];
    float rot = (t < 64) ? -xpn : xpn;
    float c = cosb[(size_t)row * HD + t];
    float s = sinb[(size_t)row * HD + t];
    float o = xn * c + rot * s;

    if (isq) {
        o *= QSCALE;
        qf[(((size_t)b * Hh + hidx) * Qn + qpos) * HD + t] = (_Float16)o;
    } else {
        kf[(((size_t)b * KVh + (hidx - Hh)) * KTOT + Mm + qpos) * HD + t] = (_Float16)o;
    }
}

// scatter V (no norm/rope) from qkv into TRANSPOSED f16 buffer [B*KV, HD, KTOT]
__global__ void k_scatter_v(const float* __restrict__ qkv, _Float16* __restrict__ vT) {
    int i = blockIdx.x * blockDim.x + threadIdx.x;  // B*KV*Q*HD
    int hd = i % HD;
    int r  = i / HD;
    int kv = r % KVh;
    int r2 = r / KVh;
    int qpos = r2 % Qn;
    int b    = r2 / Qn;
    float v = qkv[(size_t)(b * Qn + qpos) * QKVW + Hh * HD + KVh * HD + kv * HD + hd];
    vT[(((size_t)b * KVh + kv) * HD + hd) * KTOT + Mm + qpos] = (_Float16)v;
}

// copy memory prefix (f32 -> f16): K row-major, V transposed
__global__ void k_copy_mem(const float* __restrict__ mk, const float* __restrict__ mv,
                           _Float16* __restrict__ kf, _Float16* __restrict__ vT) {
    int i = blockIdx.x * blockDim.x + threadIdx.x;  // B*KV*M*HD
    int hd = i % HD;
    int t  = i / HD;
    int m  = t % Mm;
    int t2 = t / Mm;
    int kv = t2 % KVh;
    int b  = t2 / KVh;
    size_t src = (((size_t)b * KVh + kv) * Mm + m) * HD + hd;
    kf[(((size_t)b * KVh + kv) * KTOT + m) * HD + hd]  = (_Float16)mk[src];
    vT[(((size_t)b * KVh + kv) * HD + hd) * KTOT + m]  = (_Float16)mv[src];
}

// ---------------------------------------------------------------------------
// Flash attention: one wave per (b, h, 16-row q tile).
// - only the final 32-key chunk is masked (peeled)
// - row-sum of P via extra WMMA against an all-ones B matrix
// - K staged to LDS via the gfx1250 async global->LDS path when available
//   (no VGPR round trip; single s_wait_asynccnt before the barrier)
__global__ void __launch_bounds__(32)
k_flash(const _Float16* __restrict__ qf, const _Float16* __restrict__ kf,
        const _Float16* __restrict__ vT, _Float16* __restrict__ of) {
    __shared__ v16h KtV[32 * 8];        // 32 key rows x 8 groups of 16 halves
    __shared__ _Float16 Pt[16 * 32];

    const int lane = threadIdx.x;
    const int ln = lane & 15, hi = lane >> 4;
    const int qt = blockIdx.x;          // 0..Q/16-1
    const int h  = blockIdx.y;
    const int b  = blockIdx.z;
    const int kv = h >> 1;              // H/KV == 2
    const int qbase = qt * 16;

    // Q A-fragments for the 4 K-chunks of HD=128 (scale already folded in)
    v16h aq[4];
    {
        const _Float16* qp = qf + (((size_t)b * Hh + h) * Qn + qbase + ln) * HD;
        for (int dd = 0; dd < 4; ++dd) {
            h8v a0 = *(const h8v*)(qp + dd * 32 + 16 * hi);
            h8v a1 = *(const h8v*)(qp + dd * 32 + 8 + 16 * hi);
            aq[dd] = CAT16(a0, a1);
        }
    }

    v16h ones;                           // all-ones B matrix for row sums
    for (int e = 0; e < 16; ++e) ones[e] = (_Float16)1.0f;

    v8f acc[8], accl;
    for (int nt = 0; nt < 8; ++nt)
        for (int j = 0; j < 8; ++j) acc[nt][j] = 0.0f;
    for (int j = 0; j < 8; ++j) accl[j] = 0.0f;
    float mrow[8];
    for (int j = 0; j < 8; ++j) mrow[j] = -1e30f;

    const _Float16* kbase = kf + ((size_t)b * KVh + kv) * KTOT * HD;
    const _Float16* vbase = vT + ((size_t)b * KVh + kv) * HD * KTOT;

    const int totk = Mm + qbase + 16;       // last valid key index + 1
    const int nch  = (totk + 31) >> 5;      // 32-key chunks (nch >= 3)

    auto processChunk = [&](const int ks, const bool masked) {
        // stage 32 K rows into LDS (one 256B row per lane)
        {
            const _Float16* krow = kbase + (size_t)(ks + lane) * HD;
            _Float16* kp = (_Float16*)&KtV[lane * 8];
#ifdef HAVE_ASYNC_COPY
            for (int i = 0; i < 16; ++i)
                __builtin_amdgcn_global_load_async_to_lds_b128(
                    (__attribute__((address_space(1))) i32x4*)(krow + i * 8),
                    (__attribute__((address_space(3))) i32x4*)(kp + i * 8),
                    0, 0);
#else
            h8v tmp[8];
            for (int i = 0; i < 8; ++i) tmp[i] = ((const h8v*)krow)[i];
            for (int i = 0; i < 8; ++i) ((h8v*)kp)[i] = tmp[i];
            for (int i = 0; i < 8; ++i) tmp[i] = ((const h8v*)krow)[8 + i];
            for (int i = 0; i < 8; ++i) ((h8v*)kp)[8 + i] = tmp[i];
#endif
        }
        if (!masked)   // next chunk always exists in the unmasked loop
            __builtin_prefetch(kbase + (size_t)(ks + 32 + lane) * HD, 0, 1);
#ifdef HAVE_ASYNC_COPY
#if __has_builtin(__builtin_amdgcn_s_wait_asynccnt)
        __builtin_amdgcn_s_wait_asynccnt(0);
#else
        asm volatile("s_wait_asynccnt 0x0" ::: "memory");
#endif
#endif
        __syncthreads();

        // S = Q K^T for two 16-key subtiles
        v8f s[2];
        for (int c = 0; c < 2; ++c) {
            v8f sv;
            for (int j = 0; j < 8; ++j) sv[j] = 0.0f;
            for (int dd = 0; dd < 4; ++dd) {
                // contiguous 32B: key row (c*16+ln), K = dd*32 + 16*hi + e
                v16h bk = KtV[(c * 16 + ln) * 8 + dd * 2 + hi];
                sv = __builtin_amdgcn_wmma_f32_16x16x32_f16(
                    false, aq[dd], false, bk, (short)0, sv, false, false);
            }
            if (masked) {
                // valid  <=>  kpos <= Mm + qpos   (covers memory prefix too)
                int kpos = ks + c * 16 + ln;
                for (int j = 0; j < 8; ++j) {
                    int lim = Mm + qbase + j + 8 * hi;
                    if (kpos > lim) sv[j] = -1e30f;
                }
            }
            s[c] = sv;
        }

        // online softmax: row max via cross-lane tree (ds_bpermute)
        float nm[8], corr[8];
        for (int j = 0; j < 8; ++j) {
            float t = fmaxf(s[0][j], s[1][j]);
            for (int msk = 1; msk < 16; msk <<= 1)
                t = fmaxf(t, __shfl_xor(t, msk, 32));
            nm[j] = fmaxf(mrow[j], t);
            corr[j] = __expf(mrow[j] - nm[j]);
            mrow[j] = nm[j];
        }
        // exponentiate, write P tile to LDS in C layout positions
        for (int c = 0; c < 2; ++c)
            for (int j = 0; j < 8; ++j) {
                float p = __expf(s[c][j] - nm[j]);
                Pt[(j + 8 * hi) * 32 + c * 16 + ln] = (_Float16)p;
            }
        // rescale output accumulators AND the row-sum accumulator
        for (int nt = 0; nt < 8; ++nt)
            for (int j = 0; j < 8; ++j)
                acc[nt][j] *= corr[j];
        for (int j = 0; j < 8; ++j) accl[j] *= corr[j];
        __syncthreads();

        // reload P as a 16x32 A-fragment
        v16h pa;
        {
            const _Float16* pp = &Pt[ln * 32 + 16 * hi];
            h8v p0 = *(const h8v*)pp;          // e 0..7  -> k = e + 16*hi
            h8v p1 = *(const h8v*)(pp + 8);    // e 8..15 -> k = e + 16*hi
            pa = CAT16(p0, p1);
        }
        // row sums: accl += P @ Ones  (every column holds the row sum)
        accl = __builtin_amdgcn_wmma_f32_16x16x32_f16(
            false, pa, false, ones, (short)0, accl, false, false);
        // O += P V
        for (int nt = 0; nt < 8; ++nt) {
            // V B-fragment straight from global transposed buffer:
            // column d = nt*16+ln, K = ks + 16*hi + e  (contiguous 32B)
            v16h bv = *(const v16h*)(vbase + (size_t)(nt * 16 + ln) * KTOT + ks + 16 * hi);
            acc[nt] = __builtin_amdgcn_wmma_f32_16x16x32_f16(
                false, pa, false, bv, (short)0, acc[nt], false, false);
        }
        __syncthreads();
    };

    for (int ch = 0; ch < nch - 1; ++ch)
        processChunk(ch * 32, false);          // mask-free main loop
    processChunk((nch - 1) * 32, true);        // peeled causal-boundary chunk

    // normalize and write f16 output in [B,Q,H*HD] layout for the final GEMM
    float rinv[8];
    for (int j = 0; j < 8; ++j) rinv[j] = 1.0f / accl[j];
    for (int nt = 0; nt < 8; ++nt)
        for (int j = 0; j < 8; ++j) {
            int qpos = qbase + j + 8 * hi;
            float val = acc[nt][j] * rinv[j];
            of[(size_t)(b * Qn + qpos) * (Hh * HD) + h * HD + nt * 16 + ln] = (_Float16)val;
        }
}

// ---------------------------------------------------------------------------
extern "C" void kernel_launch(void* const* d_in, const int* in_sizes, int n_in,
                              void* d_out, int out_size, void* d_ws, size_t ws_size,
                              hipStream_t stream) {
    const float* hs   = (const float*)d_in[0];   // (B,Q,D)
    const float* cosb = (const float*)d_in[1];   // (B,Q,HD)
    const float* sinb = (const float*)d_in[2];
    const float* mk   = (const float*)d_in[3];   // (B,KV,M,HD)
    const float* mv   = (const float*)d_in[4];
    const float* Wq   = (const float*)d_in[5];   // (D, H*HD)
    const float* Wk   = (const float*)d_in[6];   // (D, KV*HD)
    const float* Wv   = (const float*)d_in[7];
    const float* Wo   = (const float*)d_in[8];   // (H*HD, D)
    const float* qnw  = (const float*)d_in[9];
    const float* knw  = (const float*)d_in[10];
    float* out = (float*)d_out;                  // (B,Q,D) fp32

    const int ROWS = Bb * Qn;                    // 8192

    // workspace carve-up (~295 MB, all 256B-aligned)
    char* ws = (char*)d_ws;
    _Float16* hsF   = (_Float16*)ws; ws += (size_t)ROWS * Dm * 2;
    _Float16* wqkvF = (_Float16*)ws; ws += (size_t)Dm * QKVW * 2;
    _Float16* woF   = (_Float16*)ws; ws += (size_t)(Hh*HD) * Dm * 2;
    float*    qkv   = (float*)ws;    ws += (size_t)ROWS * QKVW * 4;
    _Float16* qF    = (_Float16*)ws; ws += (size_t)Bb * Hh * Qn * HD * 2;
    _Float16* kF    = (_Float16*)ws; ws += (size_t)Bb * KVh * KTOT * HD * 2;
    _Float16* vTF   = (_Float16*)ws; ws += (size_t)Bb * KVh * HD * KTOT * 2;
    _Float16* attnF = (_Float16*)ws; ws += (size_t)ROWS * (Hh*HD) * 2;

    // 1) precision conversion / weight packing
    {
        int n = ROWS * Dm;
        k_cvt_f16<<<n / 256, 256, 0, stream>>>(hs, hsF, n);
    }
    {
        int n = Dm * QKVW;
        k_pack_wqkv<<<n / 256, 256, 0, stream>>>(Wq, Wk, Wv, wqkvF);
    }
    {
        int n = (Hh*HD) * Dm;
        k_cvt_f16<<<n / 256, 256, 0, stream>>>(Wo, woF, n);
    }

    // 2) fused QKV projection: (8192 x 2048) @ (2048 x 4096) -> fp32
    k_gemm_f16<QKVW, Dm><<<dim3(QKVW / 128, ROWS / 128), 256, 0, stream>>>(
        hsF, wqkvF, qkv);

    // 3) RMSNorm + RoPE (Q scaled), V scatter (transposed), memory prefix
    k_norm_rope<<<dim3(ROWS, Hh + KVh), 128, 0, stream>>>(
        qkv, cosb, sinb, qnw, knw, qF, kF);
    {
        int n = Bb * KVh * Qn * HD;
        k_scatter_v<<<n / 256, 256, 0, stream>>>(qkv, vTF);
    }
    {
        int n = Bb * KVh * Mm * HD;
        k_copy_mem<<<n / 256, 256, 0, stream>>>(mk, mv, kF, vTF);
    }

    // 4) flash attention (one wave per 16 query rows)
    k_flash<<<dim3(Qn / 16, Hh, Bb), 32, 0, stream>>>(qF, kF, vTF, attnF);

    // 5) output projection: (8192 x 2048) @ (2048 x 2048) -> fp32 d_out
    k_gemm_f16<Dm, Hh*HD><<<dim3(Dm / 128, ROWS / 128), 256, 0, stream>>>(
        attnF, woF, out);

    (void)in_sizes; (void)n_in; (void)out_size; (void)ws_size;
}
